// TFT5Attention_33818572489077
// MI455X (gfx1250) — compile-verified, matrix operand order checked by reading, not run on hardware
//
#include <hip/hip_runtime.h>

#define B_   4
#define S_   2048
#define D_   1024
#define H_   16
#define DK_  64
#define NBUCK 32

typedef __bf16 bf16;
typedef __attribute__((ext_vector_type(16))) __bf16 v16bf;
typedef __attribute__((ext_vector_type(8)))  __bf16 v8bf;
typedef __attribute__((ext_vector_type(8)))  float  v8f;

// ---------------------------------------------------------------------------
// Load a 16x32 bf16 fragment in the CDNA5 WMMA A-operand layout from a
// row-major matrix (row stride = ld elements).
//   lane<16 : row M=lane,   elems 0..7 = K 0..7,   elems 8..15 = K 16..23
//   lane>=16: row M=lane-16, elems 0..7 = K 8..15, elems 8..15 = K 24..31
// ---------------------------------------------------------------------------
__device__ __forceinline__ v16bf load_frag(const bf16* __restrict__ base, int ld) {
  const int lane = threadIdx.x & 31;
  const int row  = lane & 15;
  const int hi   = lane >> 4;
  const bf16* p = base + (size_t)row * ld + (hi ? 8 : 0);
  v8bf a = *(const v8bf*)(p);
  v8bf b = *(const v8bf*)(p + 16);
  v16bf f;
#pragma unroll
  for (int i = 0; i < 8; ++i) { f[i] = a[i]; f[i + 8] = b[i]; }
  return f;
}

__device__ __forceinline__ v8f wmma_bf16(v16bf a, v16bf b, v8f c) {
  return __builtin_amdgcn_wmma_f32_16x16x32_bf16(
      false, a, false, b, (short)0, c, false, false);
}

// Async copy 16B/lane from global to LDS (GLOBAL_LOAD_ASYNC_TO_LDS_B128,
// tracked by ASYNCcnt). lds_addr = LDS byte address (low 32 bits of the
// generic pointer to __shared__ memory).
__device__ __forceinline__ void async_b128(const void* gaddr, void* laddr) {
  unsigned l = (unsigned)(size_t)laddr;
  asm volatile("global_load_async_to_lds_b128 %0, %1, off"
               :: "v"(l), "v"(gaddr) : "memory");
}
__device__ __forceinline__ void wait_async0() {
  asm volatile("s_wait_asynccnt 0x0" ::: "memory");
}

// ---------------------------------------------------------------------------
__global__ void f32_to_bf16_kernel(const float* __restrict__ src,
                                   bf16* __restrict__ dst, int n) {
  int i = blockIdx.x * blockDim.x + threadIdx.x;
  if (i < n) dst[i] = (bf16)src[i];
}

__global__ void transpose_bf16_kernel(const float* __restrict__ W,
                                      bf16* __restrict__ Wt, int K, int N) {
  int i = blockIdx.x * blockDim.x + threadIdx.x;
  if (i >= K * N) return;
  int k = i / N, n = i % N;
  Wt[(size_t)n * K + k] = (bf16)W[(size_t)k * N + n];
}

// T5 relative-position bias table: tab[(delta + S-1)*H + h]
__global__ void bias_table_kernel(const float* __restrict__ rel_bias,
                                  float* __restrict__ tab) {
  int idx = blockIdx.x * blockDim.x + threadIdx.x;
  const int total = (2 * S_ - 1) * H_;
  if (idx >= total) return;
  int d = idx / H_ - (S_ - 1);
  int h = idx % H_;
  const int nb = NBUCK / 2;      // 16
  const int max_exact = nb / 2;  // 8
  int rb  = (d > 0) ? nb : 0;
  int rel = (d < 0) ? -d : d;
  int bucket;
  if (rel < max_exact) {
    bucket = rel;
  } else {
    float v = logf((float)rel / (float)max_exact) /
              logf(128.0f / (float)max_exact) * (float)(nb - max_exact);
    bucket = max_exact + (int)v;
    if (bucket > nb - 1) bucket = nb - 1;
  }
  tab[idx] = rel_bias[(rb + bucket) * H_ + h];
}

// ---------------------------------------------------------------------------
// Register-tiled WMMA GEMM: C = A(MxK) * Bt(NxK)^T
//   block = 128 threads (4 waves, 2x2), block tile 128x128, wave tile 64x64
//   (16 WMMAs per 16 b128-loads per K-step -> 2x the FLOP/byte of 32x64).
// Epilogue modes: 0/1 Q,K [B,H,S,DK] bf16; 2 V^T [B,H,DK,S] bf16; 3 f32 MxN.
// ---------------------------------------------------------------------------
__global__ __launch_bounds__(128) void gemm_bf16_kernel(
    const bf16* __restrict__ A, const bf16* __restrict__ Bt,
    void* __restrict__ Cout, int M, int N, int K, int mode) {
  const int wave = threadIdx.x >> 5;
  const int lane = threadIdx.x & 31;
  const int wm = wave & 1, wn = wave >> 1;
  const int m0 = blockIdx.x * 128 + wm * 64;
  const int n0 = blockIdx.y * 128 + wn * 64;

  v8f acc[4][4];
#pragma unroll
  for (int i = 0; i < 4; ++i)
#pragma unroll
    for (int j = 0; j < 4; ++j) acc[i][j] = {};

#pragma unroll 2
  for (int k = 0; k < K; k += 32) {
    v16bf a[4];
#pragma unroll
    for (int i = 0; i < 4; ++i)
      a[i] = load_frag(A + (size_t)(m0 + 16 * i) * K + k, K);
#pragma unroll
    for (int j = 0; j < 4; ++j) {
      v16bf bj = load_frag(Bt + (size_t)(n0 + 16 * j) * K + k, K);
#pragma unroll
      for (int i = 0; i < 4; ++i) acc[i][j] = wmma_bf16(a[i], bj, acc[i][j]);
    }
  }

  const int col = lane & 15;
#pragma unroll
  for (int i = 0; i < 4; ++i) {
#pragma unroll
    for (int j = 0; j < 4; ++j) {
#pragma unroll
      for (int r = 0; r < 8; ++r) {
        float c = acc[i][j][r];
        int m = m0 + 16 * i + ((lane < 16) ? r : r + 8);
        int n = n0 + 16 * j + col;
        if (mode == 3) {
          ((float*)Cout)[(size_t)m * N + n] = c;
        } else {
          int b = m >> 11, s = m & (S_ - 1);
          int h = n >> 6, dk = n & (DK_ - 1);
          if (mode == 2) {
            ((bf16*)Cout)[((size_t)(b * H_ + h) * DK_ + dk) * S_ + s] = (bf16)c;
          } else {
            ((bf16*)Cout)[((size_t)(b * H_ + h) * S_ + s) * DK_ + dk] = (bf16)c;
          }
        }
      }
    }
  }
}

// ---------------------------------------------------------------------------
// Flash attention. block = 256 threads (8 waves); each wave owns a 16-row
// q tile; all 8 waves share the same (b,h) K/V stream, so K/V tiles are
// staged once per block into double-buffered LDS with async-to-LDS copies
// (ASYNCcnt), cutting global K/V traffic 8x. grid = (S/128, H, B).
// Q,K: [B,H,S,DK] bf16. Vt: [B,H,DK,S] bf16. Out: [B,S,H*DK] bf16.
// ---------------------------------------------------------------------------
__global__ __launch_bounds__(256) void attn_kernel(
    const bf16* __restrict__ Q, const bf16* __restrict__ Km,
    const bf16* __restrict__ Vt, const float* __restrict__ biasTab,
    bf16* __restrict__ Out) {
  __shared__ bf16 ldsK[2][32 * 64];    // [buf][krow][d]
  __shared__ bf16 ldsV[2][64 * 32];    // [buf][d][krow]
  __shared__ bf16 lds_p[8][16 * 32];   // wave-private P tiles

  const int tid  = threadIdx.x;
  const int wave = tid >> 5;
  const int lane = tid & 31;
  const int h = blockIdx.y, b = blockIdx.z;
  const int q0 = blockIdx.x * 128 + wave * 16;
  const size_t bh = (size_t)(b * H_ + h);
  const bf16* Kbase = Km + bh * S_ * DK_;
  const bf16* Vbase = Vt + bh * (size_t)DK_ * S_;

  // cooperative async staging of one 32-key step (4KB K + 4KB V)
  auto stage = [&](int k0, int buf) {
    // K tile: rows k0..k0+31 are contiguous (32*64 bf16 = 4KB); 16B per thread
    async_b128((const char*)(Kbase + (size_t)k0 * DK_) + tid * 16,
               (char*)&ldsK[buf][0] + tid * 16);
    // V tile: 64 d-rows x 32 k; each row 64B = 4 threads x 16B
    int d = tid >> 2, qq = tid & 3;
    async_b128(Vbase + (size_t)d * S_ + k0 + qq * 8,
               &ldsV[buf][d * 32 + qq * 8]);
  };

  const bf16* qb = Q + (bh * S_ + q0) * DK_;
  v16bf aq0 = load_frag(qb, DK_);
  v16bf aq1 = load_frag(qb + 32, DK_);

  v8f o[4];
#pragma unroll
  for (int t = 0; t < 4; ++t) o[t] = {};
  float mrow[8], lrow[8];
#pragma unroll
  for (int r = 0; r < 8; ++r) { mrow[r] = -1e30f; lrow[r] = 0.0f; }

  const int colN = lane & 15;
  const int NIT = S_ / 32;

  stage(0, 0);
  for (int it = 0; it < NIT; ++it) {
    wait_async0();        // own async copies for buf[it&1] done
    __syncthreads();      // everyone's copies done; prior reads finished
    if (it + 1 < NIT) stage((it + 1) * 32, (it + 1) & 1);

    const bf16* kb = &ldsK[it & 1][0];
    const bf16* vb = &ldsV[it & 1][0];
    const int k0 = it * 32;

    v8f s0 = {}, s1 = {};
    s0 = wmma_bf16(aq0, load_frag(kb, 64), s0);
    s0 = wmma_bf16(aq1, load_frag(kb + 32, 64), s0);
    s1 = wmma_bf16(aq0, load_frag(kb + 16 * 64, 64), s1);
    s1 = wmma_bf16(aq1, load_frag(kb + 16 * 64 + 32, 64), s1);

#pragma unroll
    for (int r = 0; r < 8; ++r) {
      int row  = (lane < 16) ? r : r + 8;
      int qpos = q0 + row;
      int kpos = k0 + colN;
      float e0 = s0[r] + biasTab[(kpos - qpos + (S_ - 1)) * H_ + h];
      float e1 = s1[r] + biasTab[(kpos + 16 - qpos + (S_ - 1)) * H_ + h];

      float t = fmaxf(e0, e1);
      t = fmaxf(t, __shfl_xor(t, 1));
      t = fmaxf(t, __shfl_xor(t, 2));
      t = fmaxf(t, __shfl_xor(t, 4));
      t = fmaxf(t, __shfl_xor(t, 8));
      float mnew  = fmaxf(mrow[r], t);
      float alpha = __expf(mrow[r] - mnew);
      float p0 = __expf(e0 - mnew);
      float p1 = __expf(e1 - mnew);
      float sum = p0 + p1;
      sum += __shfl_xor(sum, 1);
      sum += __shfl_xor(sum, 2);
      sum += __shfl_xor(sum, 4);
      sum += __shfl_xor(sum, 8);
      lrow[r] = lrow[r] * alpha + sum;
      mrow[r] = mnew;
#pragma unroll
      for (int t4 = 0; t4 < 4; ++t4) o[t4][r] *= alpha;

      lds_p[wave][row * 32 + colN]      = (bf16)p0;
      lds_p[wave][row * 32 + colN + 16] = (bf16)p1;
    }

    // same-wave LDS ops are in-order (DScnt) -> no barrier for lds_p
    v16bf ap = load_frag(&lds_p[wave][0], 32);
#pragma unroll
    for (int t4 = 0; t4 < 4; ++t4)
      o[t4] = wmma_bf16(ap, load_frag(vb + t4 * 16 * 32, 32), o[t4]);
  }

#pragma unroll
  for (int r = 0; r < 8; ++r) {
    float inv = 1.0f / lrow[r];
    int row = (lane < 16) ? r : r + 8;
    size_t base = ((size_t)b * S_ + (q0 + row)) * (size_t)(H_ * DK_) + h * DK_;
#pragma unroll
    for (int t4 = 0; t4 < 4; ++t4)
      Out[base + t4 * 16 + colN] = (bf16)(o[t4][r] * inv);
  }
}

// ---------------------------------------------------------------------------
extern "C" void kernel_launch(void* const* d_in, const int* in_sizes, int n_in,
                              void* d_out, int out_size, void* d_ws, size_t ws_size,
                              hipStream_t stream) {
  (void)in_sizes; (void)n_in; (void)out_size; (void)ws_size;
  const float* hidden = (const float*)d_in[0];
  const float* Wq = (const float*)d_in[1];
  const float* Wk = (const float*)d_in[2];
  const float* Wv = (const float*)d_in[3];
  const float* Wo = (const float*)d_in[4];
  const float* relb = (const float*)d_in[5];

  char*  ws  = (char*)d_ws;
  size_t off = 0;
  auto carve = [&](size_t bytes) -> void* {
    void* p = ws + off;
    off += (bytes + 255) & ~(size_t)255;
    return p;
  };

  const size_t M = (size_t)B_ * S_;          // 8192
  bf16* hid_bf = (bf16*)carve(M * D_ * 2);
  bf16* wqT    = (bf16*)carve((size_t)D_ * D_ * 2);
  bf16* wkT    = (bf16*)carve((size_t)D_ * D_ * 2);
  bf16* wvT    = (bf16*)carve((size_t)D_ * D_ * 2);
  bf16* woT    = (bf16*)carve((size_t)D_ * D_ * 2);
  bf16* qbuf   = (bf16*)carve(M * D_ * 2);
  bf16* kbuf   = (bf16*)carve(M * D_ * 2);
  bf16* vTbuf  = (bf16*)carve(M * D_ * 2);
  bf16* attn   = (bf16*)carve(M * D_ * 2);
  float* btab  = (float*)carve((size_t)(2 * S_ - 1) * H_ * 4);

  {
    int n = (int)(M * D_);
    f32_to_bf16_kernel<<<(n + 255) / 256, 256, 0, stream>>>(hidden, hid_bf, n);
  }
  {
    int n = D_ * D_;
    dim3 g((n + 255) / 256);
    transpose_bf16_kernel<<<g, 256, 0, stream>>>(Wq, wqT, D_, D_);
    transpose_bf16_kernel<<<g, 256, 0, stream>>>(Wk, wkT, D_, D_);
    transpose_bf16_kernel<<<g, 256, 0, stream>>>(Wv, wvT, D_, D_);
    transpose_bf16_kernel<<<g, 256, 0, stream>>>(Wo, woT, D_, D_);
  }
  {
    int n = (2 * S_ - 1) * H_;
    bias_table_kernel<<<(n + 255) / 256, 256, 0, stream>>>(relb, btab);
  }
  {
    dim3 grid((unsigned)(M / 128), D_ / 128);
    gemm_bf16_kernel<<<grid, 128, 0, stream>>>(hid_bf, wqT, qbuf,  (int)M, D_, D_, 0);
    gemm_bf16_kernel<<<grid, 128, 0, stream>>>(hid_bf, wkT, kbuf,  (int)M, D_, D_, 1);
    gemm_bf16_kernel<<<grid, 128, 0, stream>>>(hid_bf, wvT, vTbuf, (int)M, D_, D_, 2);
  }
  {
    dim3 grid(S_ / 128, H_, B_);
    attn_kernel<<<grid, 256, 0, stream>>>(qbuf, kbuf, vTbuf, btab, attn);
  }
  {
    dim3 grid((unsigned)(M / 128), D_ / 128);
    gemm_bf16_kernel<<<grid, 128, 0, stream>>>(attn, woT, d_out, (int)M, D_, D_, 3);
  }
}
